// FGHGNNLayer_37941741093444
// MI455X (gfx1250) — matrix operand order repaired and checked by reading, not compiled, not run on hardware
//
#include <hip/hip_runtime.h>

// ---------------------------------------------------------------------------
// FGHGNN layer for MI455X (gfx1250, wave32).
// Dense GEMMs (GINE MLPs, GAT projections, post-GAT MLPs; ~66 GFLOP total)
// run on v_wmma_f32_16x16x32_bf16 with fp32 accumulation: 256-thread blocks,
// 128x128 tile, each of 8 waves owns a 32x64 sub-tile (8 WMMAs / K-step).
// Edge scatter / softmax paths are float-atomic based; the layer is HBM-bound
// on the 4x re-stream of edge_attr (~1.6 GB @ 23.3 TB/s => ~90us floor), so
// dense work is fused (bias+BN+ReLU+residual in the GEMM epilogue) and the
// weights are pre-transposed to bf16 [N][K] once so every GEMM global load
// is a coalesced 16B vector load.
// ---------------------------------------------------------------------------

#define NAA   50000
#define NCC   10000
#define EAA   800000
#define EA2C  50000
#define EC2A  50000
#define HDIM  128
#define PDIM  256
#define NHEAD 4
#define BNEPS 1e-5f

typedef __attribute__((ext_vector_type(16))) __bf16 v16bf;
typedef __attribute__((ext_vector_type(8)))  __bf16 v8bf;
typedef __attribute__((ext_vector_type(8)))  float  v8f;

// LDS tile row stride: 32 payload + 8 pad bf16 (80 B: 16B-aligned for b128,
// 20-bank stride -> 16 distinct banks across a half-wave's 16 rows)
#define LDSTR 40

// ---------------------------------------------------------------------------
// small utility kernels
// ---------------------------------------------------------------------------

// fp32 [K,N] row-major  ->  bf16 [N,K] row-major (coalesced GEMM B loads)
__global__ void f2bf_t_kernel(const float* __restrict__ in, __bf16* __restrict__ out,
                              int K, int N) {
    int i = blockIdx.x * blockDim.x + threadIdx.x;
    if (i >= K * N) return;
    int k = i / N, n = i % N;
    out[(size_t)n * K + k] = (__bf16)in[i];
}

__global__ void init_agg_kernel(float* __restrict__ agg, const float* __restrict__ x,
                                const float* __restrict__ eps, int k, int n) {
    int i = blockIdx.x * blockDim.x + threadIdx.x;
    if (i < n) agg[i] = (1.0f + eps[k]) * x[i];
}

__global__ void init_gat_stats_kernel(float* __restrict__ amax, float* __restrict__ denom, int n) {
    int i = blockIdx.x * blockDim.x + threadIdx.x;
    if (i < n) { amax[i] = -INFINITY; denom[i] = 0.0f; }
}

__global__ void init_gat_bias_kernel(float* __restrict__ out, const float* __restrict__ bias, int n) {
    int i = blockIdx.x * blockDim.x + threadIdx.x;   // n = n_dst * 512
    if (i < n) out[i] = bias[i & 511];
}

// ---------------------------------------------------------------------------
// GINE edge message: agg[dst] += relu(x[src] + edge_attr), float4 per thread
// ---------------------------------------------------------------------------
__global__ void gine_edge_kernel(const float* __restrict__ x, const int* __restrict__ ei,
                                 const float* __restrict__ ea, float* __restrict__ agg) {
    int idx = blockIdx.x * blockDim.x + threadIdx.x;     // E * 32 threads
    if (idx >= EAA * (HDIM / 4)) return;
    int e  = idx >> 5;
    int c4 = (idx & 31) << 2;
    int s = ei[e], d = ei[EAA + e];
    const float4 xv = *(const float4*)(x + (size_t)s * HDIM + c4);
    const float4 ev = *(const float4*)(ea + (size_t)e * HDIM + c4);
    float* o = agg + (size_t)d * HDIM + c4;
    atomicAdd(o + 0, fmaxf(xv.x + ev.x, 0.0f));
    atomicAdd(o + 1, fmaxf(xv.y + ev.y, 0.0f));
    atomicAdd(o + 2, fmaxf(xv.z + ev.z, 0.0f));
    atomicAdd(o + 3, fmaxf(xv.w + ev.w, 0.0f));
}

// ---------------------------------------------------------------------------
// GAT helpers
// ---------------------------------------------------------------------------
__device__ __forceinline__ void atomicMaxFloat(float* addr, float val) {
    if (val >= 0.0f) atomicMax((int*)addr, __float_as_int(val));
    else             atomicMin((unsigned int*)addr, __float_as_uint(val));
}

// per-(node, head) dot product with attention vector (H=128 reduce)
__global__ void attn_dot_kernel(const float* __restrict__ hfeat, const float* __restrict__ att,
                                float* __restrict__ out, int n_nodes) {
    int idx = blockIdx.x * blockDim.x + threadIdx.x;
    if (idx >= n_nodes * NHEAD) return;
    int node = idx / NHEAD, h = idx % NHEAD;
    const float4* f = (const float4*)(hfeat + (size_t)node * (NHEAD * HDIM) + h * HDIM);
    const float4* a = (const float4*)(att + h * HDIM);
    float s = 0.0f;
#pragma unroll 8
    for (int c = 0; c < HDIM / 4; ++c) {
        float4 fv = f[c], av = a[c];
        s += fv.x * av.x + fv.y * av.y + fv.z * av.z + fv.w * av.w;
    }
    out[idx] = s;
}

__global__ void edge_logit_kernel(const int* __restrict__ ei, int E,
                                  const float* __restrict__ ssrc, const float* __restrict__ sdst,
                                  float* __restrict__ elog, float* __restrict__ amax) {
    int idx = blockIdx.x * blockDim.x + threadIdx.x;
    if (idx >= E * NHEAD) return;
    int e = idx / NHEAD, h = idx % NHEAD;
    int s = ei[e], d = ei[E + e];
    float a = ssrc[s * NHEAD + h] + sdst[d * NHEAD + h];
    a = (a > 0.0f) ? a : 0.2f * a;                 // leaky_relu(0.2)
    elog[idx] = a;
    atomicMaxFloat(&amax[d * NHEAD + h], a);
}

__global__ void edge_exp_kernel(const int* __restrict__ ei, int E,
                                const float* __restrict__ elog, const float* __restrict__ amax,
                                float* __restrict__ eexp, float* __restrict__ denom) {
    int idx = blockIdx.x * blockDim.x + threadIdx.x;
    if (idx >= E * NHEAD) return;
    int e = idx / NHEAD, h = idx % NHEAD;
    int d = ei[E + e];
    float ae = __expf(elog[idx] - amax[d * NHEAD + h]);
    eexp[idx] = ae;
    atomicAdd(&denom[d * NHEAD + h], ae);
}

// out[dst, h, :] += (eexp/denom) * hs[src, h, :]   (float4 per thread)
__global__ void edge_scatter_kernel(const int* __restrict__ ei, int E,
                                    const float* __restrict__ hs,
                                    const float* __restrict__ eexp, const float* __restrict__ denom,
                                    float* __restrict__ out) {
    int idx = blockIdx.x * blockDim.x + threadIdx.x;     // E * 128 threads
    if (idx >= E * (NHEAD * HDIM / 4)) return;
    int e = idx >> 7;
    int j = (idx & 127) << 2;                            // 0..508
    int h = j >> 7;
    int s = ei[e], d = ei[E + e];
    float w = eexp[e * NHEAD + h] / denom[d * NHEAD + h];
    const float4 hv = *(const float4*)(hs + (size_t)s * (NHEAD * HDIM) + j);
    float* o = out + (size_t)d * (NHEAD * HDIM) + j;
    atomicAdd(o + 0, w * hv.x);
    atomicAdd(o + 1, w * hv.y);
    atomicAdd(o + 2, w * hv.z);
    atomicAdd(o + 3, w * hv.w);
}

// ---------------------------------------------------------------------------
// Generic WMMA GEMM: out[M,N] = epilogue(A[M,K] @ W[K,N])
//   A fp32 row-major (converted to bf16 on tile load), Wt bf16 [N][K].
//   Block: 256 threads (8 waves), 128x128 tile; wave (r,c) = (wave>>1, wave&1)
//   computes the 32x64 sub-tile [r*32, r*32+32) x [c*64, c*64+64):
//   2 A-frags x 4 B-frags = 8 WMMAs per K-step of 32.
//   Epilogue: +bias, optional eval-BN (bn = [gamma|beta|mean|var] rows of N),
//   optional ReLU, optional residual add. Requires N % 128 == 0.
// ---------------------------------------------------------------------------
__global__ __launch_bounds__(256)
void gemm_bf16_kernel(const float* __restrict__ A, const __bf16* __restrict__ Wt,
                      const float* __restrict__ bias, const float* __restrict__ bnp,
                      const float* __restrict__ residual, float* __restrict__ out,
                      int M, int N, int K, int do_relu) {
    __shared__ __bf16 As[128][LDSTR];    // M-major, padded
    __shared__ __bf16 Bs[128][LDSTR];    // N-major, padded
    const int tid  = threadIdx.x;
    const int wave = tid >> 5;
    const int lane = tid & 31;
    const int m0 = blockIdx.x * 128;
    const int n0 = blockIdx.y * 128;
    const int wm = (wave >> 1) * 32;     // wave's M offset in tile
    const int wn = (wave & 1) * 64;      // wave's N offset in tile

    const int row = tid >> 1;            // 0..127 (A row / B col loaded by this thread)
    const int cb  = (tid & 1) * 16;      // K half loaded by this thread
    const int gm  = m0 + row;
    const int gn  = n0 + row;            // always < N (N % 128 == 0)

    v8f acc[2][4] = {};

    for (int k0 = 0; k0 < K; k0 += 32) {
        // ---- A tile: 128x32 fp32 -> bf16, one row-level bounds branch ----
        {
            v8f f0 = {}, f1 = {};
            if (gm < M) {
                const float* ap = A + (size_t)gm * K + (k0 + cb);
                f0 = *(const v8f*)ap;
                f1 = *(const v8f*)(ap + 8);
                if (k0 + 32 < K) __builtin_prefetch(ap + 32, 0, 1);  // global_prefetch_b8
            }
            *(v8bf*)&As[row][cb]     = __builtin_convertvector(f0, v8bf);
            *(v8bf*)&As[row][cb + 8] = __builtin_convertvector(f1, v8bf);
        }
        // ---- B tile: Wt [N][K] bf16, contiguous 16B vector loads ----
        {
            const v8bf* wp = (const v8bf*)(Wt + (size_t)gn * K + (k0 + cb));
            *(v8bf*)&Bs[row][cb]     = wp[0];
            *(v8bf*)&Bs[row][cb + 8] = wp[1];
        }
        __syncthreads();

        // A fragments (16x32 bf16 layout): lane m = lane%16,
        // half-wave K base 0/16; K = base+2v+t
        v16bf af[2];
        {
            int akb = (lane < 16) ? 0 : 16;
#pragma unroll
            for (int mt = 0; mt < 2; ++mt) {
                int am = wm + mt * 16 + (lane & 15);
#pragma unroll
                for (int v = 0; v < 8; ++v) {
                    int kk = akb + 2 * v;
                    af[mt][2 * v]     = As[am][kk];
                    af[mt][2 * v + 1] = As[am][kk + 1];
                }
            }
        }
#pragma unroll
        for (int nt = 0; nt < 4; ++nt) {
            // B fragment (32x16): lane n = lane%16, K base 0/16, K = base+2v+t
            v16bf bf;
            int bnc = wn + nt * 16 + (lane & 15);
            int bkb = (lane < 16) ? 0 : 16;
#pragma unroll
            for (int v = 0; v < 8; ++v) {
                bf[2 * v]     = Bs[bnc][bkb + 2 * v];
                bf[2 * v + 1] = Bs[bnc][bkb + 2 * v + 1];
            }
#pragma unroll
            for (int mt = 0; mt < 2; ++mt) {
                acc[mt][nt] = __builtin_amdgcn_wmma_f32_16x16x32_bf16(
                    false, af[mt], false, bf, (short)0, acc[mt][nt], false, false);
            }
        }
        __syncthreads();
    }

    // Epilogue. C/D layout: VGPR v holds row M = v + (lane<16 ? 0 : 8), col = lane%16.
#pragma unroll
    for (int nt = 0; nt < 4; ++nt) {
        int cn = n0 + wn + nt * 16 + (lane & 15);
        float bia = bias ? bias[cn] : 0.0f;
        float g = 1.0f, bb = 0.0f, mu = 0.0f, iv = 1.0f;
        if (bnp) {
            g  = bnp[cn];
            bb = bnp[N + cn];
            mu = bnp[2 * N + cn];
            iv = rsqrtf(bnp[3 * N + cn] + BNEPS);
        }
#pragma unroll
        for (int mt = 0; mt < 2; ++mt) {
#pragma unroll
            for (int v = 0; v < 8; ++v) {
                int cm = m0 + wm + mt * 16 + v + ((lane < 16) ? 0 : 8);
                if (cm < M) {
                    float val = acc[mt][nt][v] + bia;
                    if (bnp) val = g * (val - mu) * iv + bb;
                    if (do_relu) val = fmaxf(val, 0.0f);
                    if (residual) val += residual[(size_t)cm * N + cn];
                    out[(size_t)cm * N + cn] = val;
                }
            }
        }
    }
}

// ---------------------------------------------------------------------------
// host-side orchestration
// ---------------------------------------------------------------------------
static inline int ceil_div(long long a, int b) { return (int)((a + b - 1) / b); }

static void launch_gemm(const float* A, const __bf16* Wt, const float* bias,
                        const float* bnp, const float* resid, float* out,
                        int M, int N, int K, int relu, hipStream_t s) {
    dim3 grid(ceil_div(M, 128), N / 128);
    gemm_bf16_kernel<<<grid, 256, 0, s>>>(A, Wt, bias, bnp, resid, out, M, N, K, relu);
}

static void launch_f2bf_t(const float* in, __bf16* out, int K, int N, hipStream_t s) {
    f2bf_t_kernel<<<ceil_div((long long)K * N, 256), 256, 0, s>>>(in, out, K, N);
}

extern "C" void kernel_launch(void* const* d_in, const int* in_sizes, int n_in,
                              void* d_out, int out_size, void* d_ws, size_t ws_size,
                              hipStream_t stream) {
    (void)in_sizes; (void)n_in; (void)out_size; (void)ws_size;

    // ---- inputs (setup_inputs order) ----
    const float* x_in     = (const float*)d_in[0];
    const int*   ei_aa    = (const int*)d_in[1];
    const float* ea_aa    = (const float*)d_in[2];
    const float* xcl_in   = (const float*)d_in[3];
    // d_in[4], d_in[5]: c2c graph — unused (cluster_convs empty in reference)
    const int*   ei_a2c   = (const int*)d_in[6];
    const int*   ei_c2a   = (const int*)d_in[7];
    const float* gW1      = (const float*)d_in[8];
    const float* gb1      = (const float*)d_in[9];
    const float* gbn1     = (const float*)d_in[10];
    const float* gW2      = (const float*)d_in[11];
    const float* gb2      = (const float*)d_in[12];
    const float* gbn2     = (const float*)d_in[13];
    const float* geps     = (const float*)d_in[14];
    const float* upWsrc   = (const float*)d_in[15];
    const float* upWdst   = (const float*)d_in[16];
    const float* upAsrc   = (const float*)d_in[17];
    const float* upAdst   = (const float*)d_in[18];
    const float* upBias   = (const float*)d_in[19];
    const float* plWsrc   = (const float*)d_in[20];
    const float* plWdst   = (const float*)d_in[21];
    const float* plAsrc   = (const float*)d_in[22];
    const float* plAdst   = (const float*)d_in[23];
    const float* plBias   = (const float*)d_in[24];
    const float* c2aW1    = (const float*)d_in[25];
    const float* c2ab1    = (const float*)d_in[26];
    const float* c2abn1   = (const float*)d_in[27];
    const float* c2aW2    = (const float*)d_in[28];
    const float* c2ab2    = (const float*)d_in[29];
    const float* c2abn2   = (const float*)d_in[30];
    const float* a2cW1    = (const float*)d_in[31];
    const float* a2cb1    = (const float*)d_in[32];
    const float* a2cbn1   = (const float*)d_in[33];
    const float* a2cW2    = (const float*)d_in[34];
    const float* a2cb2    = (const float*)d_in[35];
    const float* a2cbn2   = (const float*)d_in[36];

    // ---- workspace layout (fp32 elements) ----
    float* wsf = (float*)d_ws;
    size_t o = 0;
    float* xa0   = wsf + o; o += (size_t)NAA * HDIM;
    float* xa1   = wsf + o; o += (size_t)NAA * HDIM;
    float* agg   = wsf + o; o += (size_t)NAA * HDIM;
    float* hid   = wsf + o; o += (size_t)NAA * PDIM;
    float* bigA  = wsf + o; o += (size_t)NAA * 512;    // [n_atoms, NH*H]
    float* gout  = wsf + o; o += (size_t)NAA * 512;
    float* bigC  = wsf + o; o += (size_t)NCC * 512;    // [n_clusters, NH*H]
    float* ssrc  = wsf + o; o += (size_t)NAA * NHEAD;
    float* sdst  = wsf + o; o += (size_t)NAA * NHEAD;
    float* amax  = wsf + o; o += (size_t)NAA * NHEAD;
    float* denom = wsf + o; o += (size_t)NAA * NHEAD;
    float* elog  = wsf + o; o += (size_t)EA2C * NHEAD;
    float* eexp  = wsf + o; o += (size_t)EA2C * NHEAD;
    __bf16* wb = (__bf16*)(wsf + o);
    size_t bo = 0;
    __bf16* gW1b  = wb + bo; bo += 4 * HDIM * PDIM;
    __bf16* gW2b  = wb + bo; bo += 4 * PDIM * HDIM;
    __bf16* upSb  = wb + bo; bo += HDIM * 512;
    __bf16* upDb  = wb + bo; bo += HDIM * 512;
    __bf16* plSb  = wb + bo; bo += HDIM * 512;
    __bf16* plDb  = wb + bo; bo += HDIM * 512;
    __bf16* c2a1b = wb + bo; bo += 512 * PDIM;
    __bf16* c2a2b = wb + bo; bo += PDIM * HDIM;
    __bf16* a2c1b = wb + bo; bo += 512 * PDIM;
    __bf16* a2c2b = wb + bo; bo += PDIM * HDIM;

    float* out_at = (float*)d_out;
    float* out_cl = (float*)d_out + (size_t)NAA * HDIM;

    // ---- convert + transpose weights fp32 [K,N] -> bf16 [N,K] ----
    for (int k = 0; k < 4; ++k) {
        launch_f2bf_t(gW1 + (size_t)k * HDIM * PDIM, gW1b + (size_t)k * HDIM * PDIM,
                      HDIM, PDIM, stream);
        launch_f2bf_t(gW2 + (size_t)k * PDIM * HDIM, gW2b + (size_t)k * PDIM * HDIM,
                      PDIM, HDIM, stream);
    }
    launch_f2bf_t(upWsrc, upSb, HDIM, 512, stream);
    launch_f2bf_t(upWdst, upDb, HDIM, 512, stream);
    launch_f2bf_t(plWsrc, plSb, HDIM, 512, stream);
    launch_f2bf_t(plWdst, plDb, HDIM, 512, stream);
    launch_f2bf_t(c2aW1, c2a1b, 512, PDIM, stream);
    launch_f2bf_t(c2aW2, c2a2b, PDIM, HDIM, stream);
    launch_f2bf_t(a2cW1, a2c1b, 512, PDIM, stream);
    launch_f2bf_t(a2cW2, a2c2b, PDIM, HDIM, stream);

    // ---- x -> xa0 ----
    hipMemcpyAsync(xa0, x_in, (size_t)NAA * HDIM * sizeof(float),
                   hipMemcpyDeviceToDevice, stream);

    // ---- 4 GINE layers ----
    float* cur = xa0;
    float* nxt = xa1;
    for (int k = 0; k < 4; ++k) {
        int nNH = NAA * HDIM;
        init_agg_kernel<<<ceil_div(nNH, 256), 256, 0, stream>>>(agg, cur, geps, k, nNH);
        gine_edge_kernel<<<ceil_div((long long)EAA * (HDIM / 4), 256), 256, 0, stream>>>(
            cur, ei_aa, ea_aa, agg);
        launch_gemm(agg, gW1b + (size_t)k * HDIM * PDIM, gb1 + k * PDIM,
                    gbn1 + (size_t)k * 4 * PDIM, nullptr, hid,
                    NAA, PDIM, HDIM, 1, stream);
        launch_gemm(hid, gW2b + (size_t)k * PDIM * HDIM, gb2 + k * HDIM,
                    gbn2 + (size_t)k * 4 * HDIM, cur, nxt,
                    NAA, HDIM, PDIM, 1, stream);
        float* t = cur; cur = nxt; nxt = t;
    }
    const float* xfin = cur;   // post-GINE atom features

    // ==================== unpool GAT: clusters -> atoms ====================
    launch_gemm(xcl_in, upSb, nullptr, nullptr, nullptr, bigC, NCC, 512, HDIM, 0, stream); // hs
    launch_gemm(xfin,   upDb, nullptr, nullptr, nullptr, bigA, NAA, 512, HDIM, 0, stream); // hd
    attn_dot_kernel<<<ceil_div(NCC * NHEAD, 256), 256, 0, stream>>>(bigC, upAsrc, ssrc, NCC);
    attn_dot_kernel<<<ceil_div(NAA * NHEAD, 256), 256, 0, stream>>>(bigA, upAdst, sdst, NAA);
    init_gat_stats_kernel<<<ceil_div(NAA * NHEAD, 256), 256, 0, stream>>>(amax, denom, NAA * NHEAD);
    init_gat_bias_kernel<<<ceil_div((long long)NAA * 512, 256), 256, 0, stream>>>(gout, upBias, NAA * 512);
    edge_logit_kernel<<<ceil_div(EC2A * NHEAD, 256), 256, 0, stream>>>(ei_c2a, EC2A, ssrc, sdst, elog, amax);
    edge_exp_kernel<<<ceil_div(EC2A * NHEAD, 256), 256, 0, stream>>>(ei_c2a, EC2A, elog, amax, eexp, denom);
    edge_scatter_kernel<<<ceil_div((long long)EC2A * 128, 256), 256, 0, stream>>>(
        ei_c2a, EC2A, bigC, eexp, denom, gout);
    // c2atom MLP + residual(x) -> atom output
    launch_gemm(gout, c2a1b, c2ab1, c2abn1, nullptr, hid, NAA, PDIM, 512, 1, stream);
    launch_gemm(hid,  c2a2b, c2ab2, c2abn2, xfin, out_at, NAA, HDIM, PDIM, 1, stream);

    // ==================== pool GAT: atoms -> clusters ====================
    launch_gemm(xfin,   plSb, nullptr, nullptr, nullptr, bigA, NAA, 512, HDIM, 0, stream); // hs
    launch_gemm(xcl_in, plDb, nullptr, nullptr, nullptr, bigC, NCC, 512, HDIM, 0, stream); // hd
    attn_dot_kernel<<<ceil_div(NAA * NHEAD, 256), 256, 0, stream>>>(bigA, plAsrc, ssrc, NAA);
    attn_dot_kernel<<<ceil_div(NCC * NHEAD, 256), 256, 0, stream>>>(bigC, plAdst, sdst, NCC);
    init_gat_stats_kernel<<<ceil_div(NCC * NHEAD, 256), 256, 0, stream>>>(amax, denom, NCC * NHEAD);
    init_gat_bias_kernel<<<ceil_div((long long)NCC * 512, 256), 256, 0, stream>>>(gout, plBias, NCC * 512);
    edge_logit_kernel<<<ceil_div(EA2C * NHEAD, 256), 256, 0, stream>>>(ei_a2c, EA2C, ssrc, sdst, elog, amax);
    edge_exp_kernel<<<ceil_div(EA2C * NHEAD, 256), 256, 0, stream>>>(ei_a2c, EA2C, elog, amax, eexp, denom);
    edge_scatter_kernel<<<ceil_div((long long)EA2C * 128, 256), 256, 0, stream>>>(
        ei_a2c, EA2C, bigA, eexp, denom, gout);
    // atom2c MLP + residual(x_cl) -> cluster output
    launch_gemm(gout, a2c1b, a2cb1, a2cbn1, nullptr, hid, NCC, PDIM, 512, 1, stream);
    launch_gemm(hid,  a2c2b, a2cb2, a2cbn2, xcl_in, out_cl, NCC, HDIM, PDIM, 1, stream);
}
